// ChebyKANLayer_87505663689448
// MI455X (gfx1250) — matrix operand order repaired and checked by reading, not compiled, not run on hardware
//
#include <hip/hip_runtime.h>
#include <hip/hip_bf16.h>
#include <stdint.h>

// ---------------------------------------------------------------------------
// ChebyKAN as a fused-K bf16 GEMM on CDNA5 (gfx1250):
//   y = A(8192 x 9216, bf16) @ B(9216 x 1024, bf16), k = i*9 + d
//   A[n, i*9+d] = T_d(tanh(x[n,i]))     (prep kernel 1, Chebyshev recurrence)
//   B stored column-major Bws[o][k] = coeffs[i][o][d]  (prep kernel 2)
// GEMM: TDM double-buffered LDS, v_wmma_f32_16x16x32_bf16 inner loop.
// ---------------------------------------------------------------------------

typedef __attribute__((ext_vector_type(16))) __bf16 v16bf;
typedef __attribute__((ext_vector_type(8)))  float  v8f;
typedef __attribute__((ext_vector_type(4)))  unsigned int u32x4;
typedef __attribute__((ext_vector_type(4)))  int    i32x4;
typedef __attribute__((ext_vector_type(8)))  int    i32x8;

#define N_TOK   8192
#define IN_DIM  1024
#define OUT_DIM 1024
#define DEG1    9
#define KDIM    (IN_DIM * DEG1)          // 9216
#define KCHUNK  128                      // K elements per LDS stage (4 WMMA steps)
#define KCHUNKS (KDIM / KCHUNK)          // 72
#define LDS_ROW 272                      // 128*2B + 16B TDM pad -> conflict-free
#define BUF_SZ  (128 * LDS_ROW)          // 34816 B per tile buffer

// ---------------- prep 1: basis A[n][i*9+d] = T_d(tanh(x[n,i])) -------------
__global__ __launch_bounds__(256) void cheby_basis_kernel(
    const float* __restrict__ x, unsigned short* __restrict__ A) {
  int tid = blockIdx.x * 256 + threadIdx.x;        // 8192*64 threads
  int row = tid >> 6;
  int i0  = (tid & 63) << 4;                       // 16 input dims per thread
  const float* xr = x + (size_t)row * IN_DIM + i0;
  union { __bf16 h[144]; uint4 q[18]; } u;
#pragma unroll
  for (int e = 0; e < 16; ++e) {
    float t = tanhf(xr[e]);
    u.h[e * 9 + 0] = (__bf16)1.0f;
    u.h[e * 9 + 1] = (__bf16)t;
    float a = 1.0f, b = t;
#pragma unroll
    for (int d = 2; d < DEG1; ++d) {
      float c = 2.0f * t * b - a;
      u.h[e * 9 + d] = (__bf16)c;
      a = b; b = c;
    }
  }
  uint4* dst = (uint4*)(A + (size_t)row * KDIM + (size_t)i0 * DEG1);
#pragma unroll
  for (int j = 0; j < 18; ++j) dst[j] = u.q[j];
}

// ---------------- prep 2: Bws[o][i*9+d] = coeffs[i][o][d] (bf16) ------------
__global__ __launch_bounds__(256) void cheby_coef_kernel(
    const float* __restrict__ coeffs, unsigned short* __restrict__ B) {
  __shared__ unsigned short tile[32][292];         // 288 + 4 pad (ushort)
  int i0 = blockIdx.x * 32, o0 = blockIdx.y * 32;
  int t = threadIdx.x;
#pragma unroll
  for (int j = 0; j < 36; ++j) {
    int flat = t + j * 256;                        // 0..9215
    int il = flat / 288;
    int r  = flat % 288;                           // = ol*9 + d, contiguous read
    float v = coeffs[(size_t)(i0 + il) * (OUT_DIM * DEG1) + (size_t)o0 * DEG1 + r];
    tile[r / 9][il * 9 + (r % 9)] = __builtin_bit_cast(unsigned short, (__bf16)v);
  }
  __syncthreads();
  int ol = t >> 3, seg = t & 7;                    // 8 threads per o-row
  const unsigned int* src = (const unsigned int*)&tile[ol][0];
  unsigned int* dst = (unsigned int*)(B + (size_t)(o0 + ol) * KDIM + (size_t)i0 * DEG1);
#pragma unroll
  for (int j = 0; j < 18; ++j) dst[seg * 18 + j] = src[seg * 18 + j];
}

// ---------------- TDM 2D tile load: 128 rows x 128 bf16, padded into LDS ----
__device__ __forceinline__ void tdm_load_2d(uint64_t gaddr, uint32_t lds_addr,
                                            uint32_t tensor_rows) {
  u32x4 g0;
  g0.x = 1u;                                       // count=1, user descriptor
  g0.y = lds_addr;                                 // LDS byte address
  g0.z = (uint32_t)gaddr;                          // global_addr[31:0]
  g0.w = ((uint32_t)(gaddr >> 32) & 0x1FFFFFFu) | (2u << 30); // [56:32] + type=2
  i32x8 g1;
  // data_size=2B(1), pad_enable, pad_interval=64DW(code 5), pad_amount=4DW(code 3)
  g1.s0 = (int)((1u << 16) | (1u << 20) | (5u << 22) | (3u << 25));
  g1.s1 = (int)(((uint32_t)KDIM & 0xFFFFu) << 16);         // tensor_dim0 lo16
  g1.s2 = (int)(((uint32_t)KDIM >> 16) | ((tensor_rows & 0xFFFFu) << 16));
  g1.s3 = (int)((tensor_rows >> 16) | (128u << 16));       // tile_dim0 = 128
  g1.s4 = (int)128;                                        // tile_dim1 = 128, tile_dim2 = 0
  g1.s5 = (int)KDIM;                                       // tensor_dim0_stride lo32
  g1.s6 = 0;                                               // stride hi + dim1_stride lo
  g1.s7 = 0;
  i32x4 z4 = {0, 0, 0, 0};
  i32x8 z8 = {0, 0, 0, 0, 0, 0, 0, 0};
  // clang-23 / therock arity: (u32x4, i32x8, i32x4, i32x4, i32x8, i32 cpol)
  __builtin_amdgcn_tensor_load_to_lds(g0, g1, z4, z4, z8, 0);
}

// ---------------- main GEMM: 128x128 tile, 8 waves, bf16 WMMA ---------------
__global__ __launch_bounds__(256, 1) void cheby_gemm_kernel(
    const unsigned short* __restrict__ A, const unsigned short* __restrict__ B,
    float* __restrict__ out) {
  __shared__ __align__(16) char smem[4 * BUF_SZ]; // A0,A1,B0,B1 = 136 KB

  const int tid  = threadIdx.x;
  const int lane = tid & 31;
  const int wave = tid >> 5;
  const int l15  = lane & 15;
  const bool hi  = lane >= 16;
  const int wm   = (wave >> 1) * 32;              // wave M offset (0/32/64/96)
  const int wn   = (wave & 1) * 64;               // wave N offset (0/64)
  const int m0g  = blockIdx.y * 128;
  const int n0g  = blockIdx.x * 128;

  const uint32_t smemAddr = (uint32_t)(uintptr_t)&smem[0];
  const uint64_t aBaseG = (uint64_t)(uintptr_t)A + (uint64_t)m0g * (KDIM * 2);
  const uint64_t bBaseG = (uint64_t)(uintptr_t)B + (uint64_t)n0g * (KDIM * 2);

  v8f acc[2][4] = {};

  if (wave == 0) {                                 // prologue: stage chunk 0
    tdm_load_2d(aBaseG, smemAddr, N_TOK);
    tdm_load_2d(bBaseG, smemAddr + 2 * BUF_SZ, OUT_DIM);
  }

  for (int c = 0; c < KCHUNKS; ++c) {
    if (wave == 0) {
      if (c + 1 < KCHUNKS) {                       // stage chunk c+1 into other buffer
        int nb = (c + 1) & 1;
        tdm_load_2d(aBaseG + (uint64_t)(c + 1) * (KCHUNK * 2),
                    smemAddr + nb * BUF_SZ, N_TOK);
        tdm_load_2d(bBaseG + (uint64_t)(c + 1) * (KCHUNK * 2),
                    smemAddr + 2 * BUF_SZ + nb * BUF_SZ, OUT_DIM);
        __builtin_amdgcn_s_wait_tensorcnt(2);      // chunk c's two TDMs done
      } else {
        __builtin_amdgcn_s_wait_tensorcnt(0);
      }
    }
    __syncthreads();                               // chunk c visible to all waves

    const char* aB = smem + (c & 1) * BUF_SZ;
    const char* bB = smem + 2 * BUF_SZ + (c & 1) * BUF_SZ;
    const int aSel = hi ? 16 : 0;                  // A: K 8..15 in upper half-wave
    const int bSel = hi ? 16 : 0;                  // B: K 8..15 in upper half-wave

#pragma unroll
    for (int s = 0; s < 4; ++s) {                  // 4 x K=32 WMMA steps
      const int kb = s * 64;                       // 32 bf16 = 64 bytes
      union Frag { uint4 q[2]; v16bf v; };
      v16bf afr[2], bfr[4];
#pragma unroll
      for (int mi = 0; mi < 2; ++mi) {
        const char* rp = aB + (wm + mi * 16 + l15) * LDS_ROW + kb + aSel;
        Frag f;
        f.q[0] = *(const uint4*)rp;                // K 0..7   (or 8..15)
        f.q[1] = *(const uint4*)(rp + 32);         // K 16..23 (or 24..31)
        afr[mi] = f.v;
      }
#pragma unroll
      for (int ni = 0; ni < 4; ++ni) {
        const char* rp = bB + (wn + ni * 16 + l15) * LDS_ROW + kb + bSel;
        Frag f;
        f.q[0] = *(const uint4*)rp;                // K 0..7   (or 8..15)
        f.q[1] = *(const uint4*)(rp + 32);         // K 16..23 (or 24..31)
        bfr[ni] = f.v;
      }
#pragma unroll
      for (int mi = 0; mi < 2; ++mi)
#pragma unroll
        for (int ni = 0; ni < 4; ++ni)
          acc[mi][ni] = __builtin_amdgcn_wmma_f32_16x16x32_bf16(
              false, afr[mi], false, bfr[ni], (short)0, acc[mi][ni],
              false, false);
    }
    __syncthreads();                               // compute done before rewrite
  }

  // Epilogue: C layout -> VGPR r: M = r (lanes 0-15) / 8+r (lanes 16-31), N = lane&15
#pragma unroll
  for (int mi = 0; mi < 2; ++mi)
#pragma unroll
    for (int ni = 0; ni < 4; ++ni) {
      int rbase = m0g + wm + mi * 16 + (hi ? 8 : 0);
      int col   = n0g + wn + ni * 16 + l15;
#pragma unroll
      for (int r = 0; r < 8; ++r)
        out[(size_t)(rbase + r) * OUT_DIM + col] = acc[mi][ni][r];
    }
}

// ---------------------------------------------------------------------------
extern "C" void kernel_launch(void* const* d_in, const int* in_sizes, int n_in,
                              void* d_out, int out_size, void* d_ws, size_t ws_size,
                              hipStream_t stream) {
  const float* x      = (const float*)d_in[0];     // 8192 x 1024 fp32
  const float* coeffs = (const float*)d_in[1];     // 1024 x 1024 x 9 fp32
  float* out = (float*)d_out;                      // 8192 x 1024 fp32

  const size_t A_BYTES = (size_t)N_TOK * KDIM * 2; // 150,994,944
  unsigned short* Aws = (unsigned short*)d_ws;
  unsigned short* Bws = (unsigned short*)((char*)d_ws + A_BYTES);

  cheby_basis_kernel<<<(N_TOK * 64) / 256, 256, 0, stream>>>(x, Aws);
  cheby_coef_kernel<<<dim3(IN_DIM / 32, OUT_DIM / 32), 256, 0, stream>>>(coeffs, Bws);
  cheby_gemm_kernel<<<dim3(OUT_DIM / 128, N_TOK / 128), 256, 0, stream>>>(Aws, Bws, out);
}
